// EfficientMultiHeadAttention_2920577761329
// MI455X (gfx1250) — matrix operand order repaired and checked by reading, not compiled, not run on hardware
//
#include <hip/hip_runtime.h>
#include <cstdint>
#include <cstddef>

// ---------------------------------------------------------------------------
// EfficientMultiHeadAttention (linear attention) for MI455X / gfx1250.
// wave32 + V_WMMA_F32_16X16X32_BF16 everywhere a matmul appears.
// ---------------------------------------------------------------------------

#define CDIM   192
#define NTOK   16384         // H*W = 128*128
#define BATCH  4
#define NHEAD  8
#define DHEAD  24
#define TILE_N 128           // tokens per workgroup
#define MTILES 12            // 192 / 16
#define KSTEPS 6             // 192 / 32

// LDS row strides (in ushort/bf16 elements), padded for bank spread
#define XSTRIDE  200         // Xbuf: [token][C]
#define KVSTRIDE 136         // K/V buf: [channel][token]
#define YSTRIDE  200         // Ybuf: [token][E]

// workspace layout (bytes)
#define WS_W_ELEMS  (4*72*32*16)                      // 4 gemms * 72 frags * 32 lanes * 16 bf16
#define WS_CTX_OFF  (WS_W_ELEMS*2)                    // 294912
#define WS_CTX_ELEMS (BATCH*NHEAD*DHEAD*DHEAD)        // 18432 f32
#define WS_SUM_OFF  (WS_CTX_OFF + WS_CTX_ELEMS*4)     // 368640
#define WS_SUM_ELEMS (BATCH*CDIM)                     // 768 f32
#define WS_Q_OFF    372736                            // aligned up
// wsq: [b][n][256] bf16 : 8 heads * 32 slots (24 data + 8 zero pad)

typedef __attribute__((ext_vector_type(16))) __bf16 v16bf;
typedef __attribute__((ext_vector_type(8)))  float  v8f;

union Frag {
    v16bf v;
    unsigned short u[16];
    uint4 q[2];
};

__device__ __forceinline__ unsigned short f2bf(float f) {
    unsigned int u = __float_as_uint(f);
    u += 0x7FFFu + ((u >> 16) & 1u);          // round-to-nearest-even
    return (unsigned short)(u >> 16);
}
__device__ __forceinline__ float bf2f(unsigned short h) {
    return __uint_as_float(((unsigned int)h) << 16);
}

__device__ __forceinline__ v8f wmma_bf16(const Frag& a, const Frag& b, v8f c) {
    // D = A(16x32 bf16) * B(32x16 bf16) + C(16x16 f32)
    return __builtin_amdgcn_wmma_f32_16x16x32_bf16(
        /*neg_a=*/false, a.v, /*neg_b=*/false, b.v,
        /*c_mod=*/(short)0, c, /*reuse_a=*/false, /*reuse_b=*/false);
}

// ---------------------------------------------------------------------------
// Weight pre-swizzle: bake W[row][col] f32 into bf16 A-fragment layout.
// A 16x32 bf16 layout used here (contiguous k): lane<16 row M=lane, elems {K0..15};
// lane>=16 row M=lane-16, elems {K16..31}.
// ---------------------------------------------------------------------------
__global__ void __launch_bounds__(256)
prep_weights(const float* __restrict__ Wq, const float* __restrict__ Wk,
             const float* __restrict__ Wv, const float* __restrict__ Wo,
             unsigned short* __restrict__ wsW)
{
    int idx = blockIdx.x * 256 + threadIdx.x;
    if (idx >= WS_W_ELEMS) return;
    int e    = idx & 15;
    int t1   = idx >> 4;
    int lane = t1 & 31;
    int t2   = t1 >> 5;
    int f    = t2 % 72;          // mt*6 + ks
    int g    = t2 / 72;          // 0:Wq 1:Wk 2:Wv 3:Wo
    int mt = f / 6, ks = f % 6;
    int hi = lane >> 4;
    int row = mt * 16 + (lane & 15);
    int col = ks * 32 + hi * 16 + e;
    const float* W = (g == 0) ? Wq : (g == 1) ? Wk : (g == 2) ? Wv : Wo;
    wsW[idx] = f2bf(W[row * CDIM + col]);
}

// ---------------------------------------------------------------------------
// Stage one input tile x[b][c][n0..n0+127] (f32, channel-major) into LDS as
// bf16 token-major [t][c] so B-fragments are contiguous 32B reads.
// ---------------------------------------------------------------------------
__device__ __forceinline__ void stage_x(const float* __restrict__ xg, int n0,
                                        unsigned short* xbuf, int tid)
{
    for (int idx = tid; idx < CDIM * TILE_N; idx += 256) {
        int t = idx & 127;
        int c = idx >> 7;
        xbuf[t * XSTRIDE + c] = f2bf(xg[(size_t)c * NTOK + n0 + t]);
    }
}

// ---------------------------------------------------------------------------
// One 192x128 projection per workgroup; wave `nt` owns a 16-token column and
// all 12 output row tiles (B-fragments loaded once, reused over 12 M-tiles).
// MODE 0: exp(z+bias) -> kvout (bf16)   MODE 1: z+bias -> kvout (bf16)
// MODE 2: z+bias -> qlog (f32)
// ---------------------------------------------------------------------------
template<int MODE>
__device__ __forceinline__ void proj_gemm(const unsigned short* __restrict__ wsWg,
                                          const float* __restrict__ bias,
                                          const unsigned short* xbuf,
                                          unsigned short* kvout, float* qlog,
                                          int nt, int lane)
{
    int lane15 = lane & 15;
    int hi     = lane >> 4;
    int tcol   = nt * 16 + lane15;

    Frag bfrag[KSTEPS];
    #pragma unroll
    for (int ks = 0; ks < KSTEPS; ++ks) {
        const unsigned short* p = xbuf + tcol * XSTRIDE + ks * 32 + hi * 16;
        bfrag[ks].q[0] = *(const uint4*)p;
        bfrag[ks].q[1] = *(const uint4*)(p + 8);
    }
    #pragma unroll
    for (int mt = 0; mt < MTILES; ++mt) {
        v8f acc = {0.f, 0.f, 0.f, 0.f, 0.f, 0.f, 0.f, 0.f};
        #pragma unroll
        for (int ks = 0; ks < KSTEPS; ++ks) {
            Frag a;
            const uint4* ap = (const uint4*)(wsWg + ((mt * 6 + ks) * 32 + lane) * 16);
            a.q[0] = ap[0];
            a.q[1] = ap[1];
            acc = wmma_bf16(a, bfrag[ks], acc);
        }
        #pragma unroll
        for (int r = 0; r < 8; ++r) {
            int row = mt * 16 + hi * 8 + r;   // D layout: lane<16 M=r, lane>=16 M=8+r
            float z = acc[r] + bias[row];
            if (MODE == 0)      kvout[row * KVSTRIDE + tcol] = f2bf(__expf(z));
            else if (MODE == 1) kvout[row * KVSTRIDE + tcol] = f2bf(z);
            else                qlog[row * TILE_N + tcol] = z;
        }
    }
}

// ---------------------------------------------------------------------------
// Fused pass 1: K-proj -> exp -> (sumexp, ctx partial via WMMA), V-proj,
// Q-proj -> per-head softmax over d -> wsq (token-major, zero-padded heads).
// LDS: Xbuf 51200B | Kbuf 52224B | Vbuf 52224B  (Qlog f32 overlays K+V)
// ---------------------------------------------------------------------------
__global__ void __launch_bounds__(256)
fused_kv_q_kernel(const float* __restrict__ q, const float* __restrict__ k,
                  const float* __restrict__ v,
                  const float* __restrict__ bq, const float* __restrict__ bk,
                  const float* __restrict__ bv,
                  const unsigned short* __restrict__ wsW,
                  float* __restrict__ wsctx, float* __restrict__ wssum,
                  unsigned short* __restrict__ wsq)
{
    extern __shared__ char smem[];
    unsigned short* Xbuf = (unsigned short*)smem;                    // 128*200 ushort
    unsigned short* Kbuf = (unsigned short*)(smem + 51200);          // 192*136 ushort
    unsigned short* Vbuf = (unsigned short*)(smem + 51200 + 52224);  // 192*136 ushort
    float*          Qlog = (float*)(smem + 51200);                   // 192*128 f32 (overlay)

    int tid  = threadIdx.x;
    int lane = tid & 31;
    int wave = tid >> 5;                 // 8 waves, wave == token-subcolumn nt
    int b    = blockIdx.y;
    int n0   = blockIdx.x * TILE_N;
    size_t xbase = (size_t)b * CDIM * NTOK;

    // ---- K projection -> exp ----
    stage_x(k + xbase, n0, Xbuf, tid);
    __syncthreads();
    proj_gemm<0>(wsW + 1 * 72 * 512, bk, Xbuf, Kbuf, nullptr, wave, lane);
    __syncthreads();

    // ---- sumexp rows (global softmax-over-N denominator) + stage V ----
    if (tid < CDIM) {
        const uint4* rp = (const uint4*)(Kbuf + tid * KVSTRIDE);
        float s = 0.f;
        #pragma unroll
        for (int j = 0; j < 16; ++j) {            // 16 * 8 bf16 = 128 tokens
            uint4 d4 = rp[j];
            unsigned w0 = d4.x, w1 = d4.y, w2 = d4.z, w3 = d4.w;
            s += bf2f((unsigned short)(w0 & 0xFFFF)) + bf2f((unsigned short)(w0 >> 16));
            s += bf2f((unsigned short)(w1 & 0xFFFF)) + bf2f((unsigned short)(w1 >> 16));
            s += bf2f((unsigned short)(w2 & 0xFFFF)) + bf2f((unsigned short)(w2 >> 16));
            s += bf2f((unsigned short)(w3 & 0xFFFF)) + bf2f((unsigned short)(w3 >> 16));
        }
        atomicAdd(&wssum[b * CDIM + tid], s);
    }
    stage_x(v + xbase, n0, Xbuf, tid);
    __syncthreads();

    // ---- V projection ----
    proj_gemm<1>(wsW + 2 * 72 * 512, bv, Xbuf, Vbuf, nullptr, wave, lane);
    __syncthreads();

    // ---- ctx[d][e] += sum_n expK[d,n] * V[e,n]  (wave == head, WMMA) ----
    {
        int h = wave;
        int lane15 = lane & 15, hi = lane >> 4;
        #pragma unroll
        for (int mt2 = 0; mt2 < 2; ++mt2) {      // d-row tiles {0..15},{16..23 pad}
            Frag afr[4];
            int drow = mt2 * 16 + lane15;
            #pragma unroll
            for (int ks = 0; ks < 4; ++ks) {     // 128 tokens = 4 * 32
                if (drow < DHEAD) {
                    const unsigned short* p =
                        Kbuf + (h * DHEAD + drow) * KVSTRIDE + ks * 32 + hi * 16;
                    afr[ks].q[0] = *(const uint4*)p;
                    afr[ks].q[1] = *(const uint4*)(p + 8);
                } else {
                    afr[ks].q[0] = make_uint4(0, 0, 0, 0);
                    afr[ks].q[1] = make_uint4(0, 0, 0, 0);
                }
            }
            #pragma unroll
            for (int nt2 = 0; nt2 < 2; ++nt2) {  // e-col tiles {0..15},{16..23 pad}
                int e = nt2 * 16 + lane15;
                v8f acc = {0.f, 0.f, 0.f, 0.f, 0.f, 0.f, 0.f, 0.f};
                #pragma unroll
                for (int ks = 0; ks < 4; ++ks) {
                    Frag bfr;
                    if (e < DHEAD) {
                        const unsigned short* p =
                            Vbuf + (h * DHEAD + e) * KVSTRIDE + ks * 32 + hi * 16;
                        bfr.q[0] = *(const uint4*)p;
                        bfr.q[1] = *(const uint4*)(p + 8);
                    } else {
                        bfr.q[0] = make_uint4(0, 0, 0, 0);
                        bfr.q[1] = make_uint4(0, 0, 0, 0);
                    }
                    acc = wmma_bf16(afr[ks], bfr, acc);
                }
                if (e < DHEAD) {
                    #pragma unroll
                    for (int r = 0; r < 8; ++r) {
                        int d = mt2 * 16 + hi * 8 + r;
                        if (d < DHEAD)
                            atomicAdd(&wsctx[(((size_t)b * NHEAD + h) * DHEAD + d) * DHEAD + e],
                                      acc[r]);
                    }
                }
            }
        }
    }
    __syncthreads();   // Kbuf/Vbuf reads done -> Qlog overlay is now safe

    // ---- Q projection -> per-token softmax over d ----
    stage_x(q + xbase, n0, Xbuf, tid);
    __syncthreads();
    proj_gemm<2>(wsW + 0, bq, Xbuf, nullptr, Qlog, wave, lane);
    __syncthreads();

    {
        int t = tid & 127;
        for (int h = tid >> 7; h < NHEAD; h += 2) {
            float vals[DHEAD];
            float m = -1e30f;
            #pragma unroll
            for (int d = 0; d < DHEAD; ++d) {
                float x = Qlog[(h * DHEAD + d) * TILE_N + t];
                vals[d] = x;
                m = fmaxf(m, x);
            }
            float s = 0.f;
            #pragma unroll
            for (int d = 0; d < DHEAD; ++d) { vals[d] = __expf(vals[d] - m); s += vals[d]; }
            float inv = 1.f / s;
            unsigned int pk[16];
            #pragma unroll
            for (int j = 0; j < 12; ++j)
                pk[j] = (unsigned)f2bf(vals[2*j] * inv) | ((unsigned)f2bf(vals[2*j+1] * inv) << 16);
            #pragma unroll
            for (int j = 12; j < 16; ++j) pk[j] = 0;   // zero pad slots d=24..31
            uint4* dst = (uint4*)(wsq + ((size_t)b * NTOK + n0 + t) * 256 + h * 32);
            dst[0] = make_uint4(pk[0],  pk[1],  pk[2],  pk[3]);
            dst[1] = make_uint4(pk[4],  pk[5],  pk[6],  pk[7]);
            dst[2] = make_uint4(pk[8],  pk[9],  pk[10], pk[11]);
            dst[3] = make_uint4(pk[12], pk[13], pk[14], pk[15]);
        }
    }
}

// ---------------------------------------------------------------------------
// Pass 2: normalize ctx, y = ctx^T @ Qsoft per head (WMMA), then out = Wo@y+bo.
// LDS: Ctxn 12288B (bf16 [h][e][32d]) | Ybuf 51200B (bf16 [t][192e] padded)
// ---------------------------------------------------------------------------
__global__ void __launch_bounds__(256)
out_kernel(const float* __restrict__ wsctx, const float* __restrict__ wssum,
           const unsigned short* __restrict__ wsq,
           const unsigned short* __restrict__ wsW,
           const float* __restrict__ bo, float* __restrict__ out)
{
    extern __shared__ char smem[];
    unsigned short* Ctxn = (unsigned short*)smem;             // 8*24*32 ushort
    unsigned short* Ybuf = (unsigned short*)(smem + 12288);   // 128*200 ushort

    int tid  = threadIdx.x;
    int lane = tid & 31;
    int wave = tid >> 5;
    int b    = blockIdx.y;
    int n0   = blockIdx.x * TILE_N;

    // ---- normalized context into A-fragment-friendly LDS: [h][e][d(32,pad0)] ----
    for (int idx = tid; idx < NHEAD * DHEAD * 32; idx += 256) {
        int d    = idx & 31;
        int rest = idx >> 5;
        int e    = rest % DHEAD;
        int h    = rest / DHEAD;
        float val = 0.f;
        if (d < DHEAD) {
            float c = wsctx[(((size_t)b * NHEAD + h) * DHEAD + d) * DHEAD + e];
            float s = wssum[b * CDIM + h * DHEAD + d];
            val = c / s;
        }
        Ctxn[(h * DHEAD + e) * 32 + d] = f2bf(val);
    }
    __syncthreads();

    // ---- y[e,t] = sum_d ctxn[d,e]*q[d,t] per head; store transposed in Ybuf ----
    {
        int nt = wave, lane15 = lane & 15, hi = lane >> 4;
        int tloc = nt * 16 + lane15;
        #pragma unroll
        for (int h = 0; h < NHEAD; ++h) {
            Frag bfr;   // B: [K=d(32) x N=16 tokens], d-pad slots are zero in wsq
            const unsigned short* p =
                wsq + ((size_t)b * NTOK + n0 + tloc) * 256 + h * 32 + hi * 16;
            bfr.q[0] = *(const uint4*)p;
            bfr.q[1] = *(const uint4*)(p + 8);
            #pragma unroll
            for (int mt2 = 0; mt2 < 2; ++mt2) {
                Frag a;
                int erow = mt2 * 16 + lane15;
                if (erow < DHEAD) {
                    const unsigned short* ap = Ctxn + (h * DHEAD + erow) * 32 + hi * 16;
                    a.q[0] = *(const uint4*)ap;
                    a.q[1] = *(const uint4*)(ap + 8);
                } else {
                    a.q[0] = make_uint4(0, 0, 0, 0);
                    a.q[1] = make_uint4(0, 0, 0, 0);
                }
                v8f acc = {0.f, 0.f, 0.f, 0.f, 0.f, 0.f, 0.f, 0.f};
                acc = wmma_bf16(a, bfr, acc);
                #pragma unroll
                for (int r = 0; r < 8; ++r) {
                    int ee = mt2 * 16 + hi * 8 + r;
                    if (ee < DHEAD)
                        Ybuf[tloc * YSTRIDE + h * DHEAD + ee] = f2bf(acc[r]);
                }
            }
        }
    }
    __syncthreads();

    // ---- out = Wo @ y + bo ----
    {
        int nt = wave, lane15 = lane & 15, hi = lane >> 4;
        int tloc = nt * 16 + lane15;
        Frag bfrag[KSTEPS];
        #pragma unroll
        for (int ks = 0; ks < KSTEPS; ++ks) {
            const unsigned short* p = Ybuf + tloc * YSTRIDE + ks * 32 + hi * 16;
            bfrag[ks].q[0] = *(const uint4*)p;
            bfrag[ks].q[1] = *(const uint4*)(p + 8);
        }
        const unsigned short* wsWo = wsW + 3 * 72 * 512;
        #pragma unroll
        for (int mt = 0; mt < MTILES; ++mt) {
            v8f acc = {0.f, 0.f, 0.f, 0.f, 0.f, 0.f, 0.f, 0.f};
            #pragma unroll
            for (int ks = 0; ks < KSTEPS; ++ks) {
                Frag a;
                const uint4* ap = (const uint4*)(wsWo + ((mt * 6 + ks) * 32 + lane) * 16);
                a.q[0] = ap[0];
                a.q[1] = ap[1];
                acc = wmma_bf16(a, bfrag[ks], acc);
            }
            #pragma unroll
            for (int r = 0; r < 8; ++r) {
                int o = mt * 16 + hi * 8 + r;
                out[((size_t)b * CDIM + o) * NTOK + n0 + tloc] = acc[r] + bo[o];
            }
        }
    }
}

// ---------------------------------------------------------------------------
extern "C" void kernel_launch(void* const* d_in, const int* in_sizes, int n_in,
                              void* d_out, int out_size, void* d_ws, size_t ws_size,
                              hipStream_t stream)
{
    const float* q  = (const float*)d_in[0];
    const float* k  = (const float*)d_in[1];
    const float* v  = (const float*)d_in[2];
    const float* Wq = (const float*)d_in[3];
    const float* bq = (const float*)d_in[4];
    const float* Wk = (const float*)d_in[5];
    const float* bk = (const float*)d_in[6];
    const float* Wv = (const float*)d_in[7];
    const float* bv = (const float*)d_in[8];
    const float* Wo = (const float*)d_in[9];
    const float* bo = (const float*)d_in[10];
    float* out = (float*)d_out;

    char* ws = (char*)d_ws;
    unsigned short* wsW   = (unsigned short*)ws;
    float*          wsctx = (float*)(ws + WS_CTX_OFF);
    float*          wssum = (float*)(ws + WS_SUM_OFF);
    unsigned short* wsq   = (unsigned short*)(ws + WS_Q_OFF);

    // zero the global accumulators (ctx + sumexp are contiguous)
    hipMemsetAsync(ws + WS_CTX_OFF, 0, (size_t)(WS_CTX_ELEMS + WS_SUM_ELEMS) * sizeof(float),
                   stream);

    prep_weights<<<(WS_W_ELEMS + 255) / 256, 256, 0, stream>>>(Wq, Wk, Wv, Wo, wsW);

    const int k1_lds = 51200 + 52224 + 52224;   // 155648 B (< 320KB WGP LDS)
    hipFuncSetAttribute((const void*)fused_kv_q_kernel,
                        hipFuncAttributeMaxDynamicSharedMemorySize, k1_lds);
    const int k3_lds = 12288 + 51200;           // 63488 B
    hipFuncSetAttribute((const void*)out_kernel,
                        hipFuncAttributeMaxDynamicSharedMemorySize, k3_lds);

    dim3 grid(NTOK / TILE_N, BATCH);            // 128 x 4 workgroups
    fused_kv_q_kernel<<<grid, 256, k1_lds, stream>>>(q, k, v, bq, bk, bv,
                                                     wsW, wsctx, wssum, wsq);
    out_kernel<<<grid, 256, k3_lds, stream>>>(wsctx, wssum, wsq, wsW, bo, out);
}
